// Llama4TextExperts_45251775431031
// MI455X (gfx1250) — matrix operand order repaired and checked by reading, not compiled, not run on hardware
//
#include <hip/hip_runtime.h>

// Llama4 MoE experts on gfx1250 (MI455X, wave32).
// Fast path: one-time f32->bf16 convert (+transpose for weights) into ws,
// then two double-buffered bf16 WMMA GEMM kernels whose staging is pure
// async global->LDS 16B copies (no per-iteration conversion VALU).
// Fallback (small ws): inline-conversion GEMM kernels.

#define E_NUM 8
#define HID   2048
#define INTER 4096
#define TPE   1024   // tokens per expert

typedef __attribute__((ext_vector_type(16))) __bf16 v16bf;
typedef __attribute__((ext_vector_type(8)))  __bf16 v8bf;
typedef __attribute__((ext_vector_type(8)))  float  v8f;

#define HAS_ASYNC_LDS (__has_builtin(__builtin_amdgcn_global_load_async_to_lds_b128) && \
                       __has_builtin(__builtin_amdgcn_s_wait_asynccnt))

#if HAS_ASYNC_LDS
typedef int v4i_t __attribute__((__vector_size__(16)));
typedef __attribute__((address_space(1))) v4i_t* g4p_t;   // global int4*
typedef __attribute__((address_space(3))) v4i_t* l4p_t;   // LDS int4*
#endif

__device__ __forceinline__ unsigned short f2bf(float f) {
  unsigned int u = __float_as_uint(f);
  u += 0x7fffu + ((u >> 16) & 1u);     // round-to-nearest-even
  return (unsigned short)(u >> 16);
}

__device__ __forceinline__ v16bf cat8(v8bf lo, v8bf hi) {
  return __builtin_shufflevector(lo, hi, 0,1,2,3,4,5,6,7,8,9,10,11,12,13,14,15);
}

__device__ __forceinline__ float silu(float g) {
  return g * __builtin_amdgcn_rcpf(1.0f + __expf(-g));
}

// 16-byte global -> LDS copy: async DMA path on CDNA5, plain copy otherwise.
__device__ __forceinline__ void copy16_g2l(const unsigned short* g, unsigned short* l) {
#if HAS_ASYNC_LDS
  __builtin_amdgcn_global_load_async_to_lds_b128((g4p_t)(g), (l4p_t)(l), 0, 0);
#else
  *(uint4*)l = *(const uint4*)g;
#endif
}
__device__ __forceinline__ void copy_fence() {
#if HAS_ASYNC_LDS
  __builtin_amdgcn_s_wait_asynccnt(0);   // s_wait_asynccnt 0
#endif
}

// ---------------------------------------------------------------------------
// Pre-pass A: flat f32 -> bf16 (X)
// ---------------------------------------------------------------------------
__global__ __launch_bounds__(256)
void cvt_flat(const float* __restrict__ src, unsigned short* __restrict__ dst) {
  long i = (long)blockIdx.x * 256 + threadIdx.x;   // one 8-elem chunk per thread
  const float4* s = (const float4*)(src + i * 8);
  float4 a = s[0], b = s[1];
  unsigned int p0 = (unsigned)f2bf(a.x) | ((unsigned)f2bf(a.y) << 16);
  unsigned int p1 = (unsigned)f2bf(a.z) | ((unsigned)f2bf(a.w) << 16);
  unsigned int p2 = (unsigned)f2bf(b.x) | ((unsigned)f2bf(b.y) << 16);
  unsigned int p3 = (unsigned)f2bf(b.z) | ((unsigned)f2bf(b.w) << 16);
  *(uint4*)(dst + i * 8) = make_uint4(p0, p1, p2, p3);
}

// ---------------------------------------------------------------------------
// Pre-pass B: per-expert transpose+convert. src: K x N f32 row-major.
// dst: N x K bf16 row-major (so GEMM B tiles are contiguous [n][k]).
// Tile 64k x 64n via padded LDS.
// ---------------------------------------------------------------------------
__global__ __launch_bounds__(256)
void cvt_transpose(const float* __restrict__ src, unsigned short* __restrict__ dst,
                   int K, int N) {
  __shared__ alignas(16) unsigned short L[64 * 72];   // padded rows (16B aligned)
  const int e  = blockIdx.z;
  const float* S = src + (size_t)e * K * N;
  unsigned short* D = dst + (size_t)e * K * N;
  const int n0 = blockIdx.x * 64, k0 = blockIdx.y * 64;
  const int tid = threadIdx.x;
  const int n = tid & 63, kt0 = tid >> 6;
#pragma unroll
  for (int j = 0; j < 16; ++j) {
    int kt = kt0 + 4 * j;
    L[n * 72 + kt] = f2bf(S[(size_t)(k0 + kt) * N + n0 + n]);  // coalesced reads
  }
  __syncthreads();
  const int f = tid >> 2, kc = (tid & 3) * 16;
  uint4* dp = (uint4*)&D[(size_t)(n0 + f) * K + k0 + kc];
  const uint4* sp = (const uint4*)&L[f * 72 + kc];
  dp[0] = sp[0];
  dp[1] = sp[1];
}

// ---------------------------------------------------------------------------
// Fast kernel 1: acted = up*silu(gate); A=Xb (bf16 [t][k]), B=W1t (bf16 [f][k]).
// Double-buffered async staging; per wave 32x32 gate + 32x32 up; 8 wmma/K-step.
// ---------------------------------------------------------------------------
__global__ __launch_bounds__(256)
void k1_fast(const unsigned short* __restrict__ Xb,
             const unsigned short* __restrict__ W1t,
             unsigned short* __restrict__ acted) {
  constexpr int BM = 128, BNA = 64, KT = 32, NK = HID / KT;
  __shared__ alignas(32) unsigned short Al[2][BM * KT];
  __shared__ alignas(32) unsigned short Bl[2][128 * KT];

  const int e = blockIdx.z, mb = blockIdx.y * BM, f0 = blockIdx.x * BNA;
  const int tid = threadIdx.x, lane = tid & 31, wid = tid >> 5;
  const int wm = (wid & 3) * 32, wn = (wid >> 2) * 32;

  const unsigned short* Ae = Xb + (size_t)(e * TPE + mb) * HID;
  const unsigned short* Be = W1t + (size_t)e * (2 * INTER) * HID;

  const v8f vzero = {0.f, 0.f, 0.f, 0.f, 0.f, 0.f, 0.f, 0.f};
  v8f accG[2][2], accU[2][2];
#pragma unroll
  for (int mt = 0; mt < 2; ++mt)
#pragma unroll
    for (int nt = 0; nt < 2; ++nt) { accG[mt][nt] = vzero; accU[mt][nt] = vzero; }

#define K1_STAGE(buf, k0)                                                      \
  {                                                                            \
    _Pragma("unroll")                                                          \
    for (int i = 0; i < 2; ++i) {                                              \
      int c = tid + i * 256, row = c >> 2, kc = (c & 3) * 8;                   \
      copy16_g2l(Ae + (size_t)row * HID + (k0) + kc, &Al[buf][row * KT + kc]); \
    }                                                                          \
    _Pragma("unroll")                                                          \
    for (int i = 0; i < 2; ++i) {                                              \
      int c = tid + i * 256, n = c >> 2, kc = (c & 3) * 8;                     \
      int fr = (n < BNA) ? (f0 + n) : (INTER + f0 + (n - BNA));                \
      copy16_g2l(Be + (size_t)fr * HID + (k0) + kc, &Bl[buf][n * KT + kc]);    \
    }                                                                          \
  }

  K1_STAGE(0, 0)
  for (int ks = 0; ks < NK; ++ks) {
    const int buf = ks & 1;
    copy_fence();
    __syncthreads();
    if (ks + 1 < NK) K1_STAGE(buf ^ 1, (ks + 1) * KT)

    v16bf af[2];
    {
      int r15 = lane & 15, kb = (lane >> 4) * 16;
#pragma unroll
      for (int mt = 0; mt < 2; ++mt) {
        const unsigned short* p = &Al[buf][(wm + mt * 16 + r15) * KT + kb];
        af[mt] = cat8(*(const v8bf*)p, *(const v8bf*)(p + 8));
      }
    }
    const int kb2 = (lane >> 4) * 16;
#pragma unroll
    for (int nt = 0; nt < 2; ++nt) {
      int n = wn + nt * 16 + (lane & 15);
      v16bf bg = *(const v16bf*)&Bl[buf][n * KT + kb2];
      v16bf bu = *(const v16bf*)&Bl[buf][(BNA + n) * KT + kb2];
#pragma unroll
      for (int mt = 0; mt < 2; ++mt) {
        accG[mt][nt] = __builtin_amdgcn_wmma_f32_16x16x32_bf16(
            false, af[mt], false, bg, (short)0, accG[mt][nt], false, false);
        accU[mt][nt] = __builtin_amdgcn_wmma_f32_16x16x32_bf16(
            false, af[mt], false, bu, (short)0, accU[mt][nt], false, false);
      }
    }
  }
#undef K1_STAGE

  unsigned short* Oe = acted + (size_t)e * TPE * INTER;
  const int rofs = (lane >> 4) << 3, c15 = lane & 15;
#pragma unroll
  for (int mt = 0; mt < 2; ++mt)
#pragma unroll
    for (int nt = 0; nt < 2; ++nt) {
      int row0 = mb + wm + mt * 16 + rofs;
      int col  = f0 + wn + nt * 16 + c15;
#pragma unroll
      for (int v = 0; v < 8; ++v)
        Oe[(size_t)(row0 + v) * INTER + col] =
            f2bf(accU[mt][nt][v] * silu(accG[mt][nt][v]));
    }
}

// ---------------------------------------------------------------------------
// Fast kernel 2: out = acted @ W2;  A=acted (bf16 [t][k]), B=W2t (bf16 [h][k]).
// ---------------------------------------------------------------------------
__global__ __launch_bounds__(256)
void k2_fast(const unsigned short* __restrict__ acted,
             const unsigned short* __restrict__ W2t,
             float* __restrict__ out) {
  constexpr int BM = 128, BN = 64, KT = 32, NK = INTER / KT;
  __shared__ alignas(32) unsigned short Al[2][BM * KT];
  __shared__ alignas(32) unsigned short Bl[2][BN * KT];

  const int e = blockIdx.z, mb = blockIdx.y * BM, n0 = blockIdx.x * BN;
  const int tid = threadIdx.x, lane = tid & 31, wid = tid >> 5;
  const int wm = (wid & 3) * 32, wn = (wid >> 2) * 32;

  const unsigned short* Ae = acted + (size_t)(e * TPE + mb) * INTER;
  const unsigned short* Be = W2t + (size_t)e * HID * INTER;

  const v8f vzero = {0.f, 0.f, 0.f, 0.f, 0.f, 0.f, 0.f, 0.f};
  v8f acc[2][2];
#pragma unroll
  for (int mt = 0; mt < 2; ++mt)
#pragma unroll
    for (int nt = 0; nt < 2; ++nt) acc[mt][nt] = vzero;

#define K2_STAGE(buf, k0)                                                       \
  {                                                                             \
    _Pragma("unroll")                                                           \
    for (int i = 0; i < 2; ++i) {                                               \
      int c = tid + i * 256, row = c >> 2, kc = (c & 3) * 8;                    \
      copy16_g2l(Ae + (size_t)row * INTER + (k0) + kc, &Al[buf][row * KT + kc]);\
    }                                                                           \
    {                                                                           \
      int n = tid >> 2, kc = (tid & 3) * 8;                                     \
      copy16_g2l(Be + (size_t)(n0 + n) * INTER + (k0) + kc,                     \
                 &Bl[buf][n * KT + kc]);                                        \
    }                                                                           \
  }

  K2_STAGE(0, 0)
  for (int ks = 0; ks < NK; ++ks) {
    const int buf = ks & 1;
    copy_fence();
    __syncthreads();
    if (ks + 1 < NK) K2_STAGE(buf ^ 1, (ks + 1) * KT)

    v16bf af[2];
    {
      int r15 = lane & 15, kb = (lane >> 4) * 16;
#pragma unroll
      for (int mt = 0; mt < 2; ++mt) {
        const unsigned short* p = &Al[buf][(wm + mt * 16 + r15) * KT + kb];
        af[mt] = cat8(*(const v8bf*)p, *(const v8bf*)(p + 8));
      }
    }
    const int kb2 = (lane >> 4) * 16;
#pragma unroll
    for (int nt = 0; nt < 2; ++nt) {
      int n = wn + nt * 16 + (lane & 15);
      v16bf b = *(const v16bf*)&Bl[buf][n * KT + kb2];
#pragma unroll
      for (int mt = 0; mt < 2; ++mt)
        acc[mt][nt] = __builtin_amdgcn_wmma_f32_16x16x32_bf16(
            false, af[mt], false, b, (short)0, acc[mt][nt], false, false);
    }
  }
#undef K2_STAGE

  float* Oe = out + (size_t)e * TPE * HID;
  const int rofs = (lane >> 4) << 3, c15 = lane & 15;
#pragma unroll
  for (int mt = 0; mt < 2; ++mt)
#pragma unroll
    for (int nt = 0; nt < 2; ++nt) {
      int row0 = mb + wm + mt * 16 + rofs;
      int col  = n0 + wn + nt * 16 + c15;
#pragma unroll
      for (int v = 0; v < 8; ++v)
        Oe[(size_t)(row0 + v) * HID + col] = acc[mt][nt][v];
    }
}

// ---------------------------------------------------------------------------
// Legacy (small-ws) kernels: inline f32->bf16 conversion in the staging loop.
// ---------------------------------------------------------------------------
__global__ __launch_bounds__(256)
void k1_legacy(const float* __restrict__ X, const float* __restrict__ W1,
               unsigned short* __restrict__ acted) {
  constexpr int BM = 128, BNA = 64, KT = 32;
  __shared__ alignas(32) unsigned short Al[BM * KT];
  __shared__ alignas(32) unsigned short Bl[128 * KT];
  const int e = blockIdx.z, mb = blockIdx.y * BM, f0 = blockIdx.x * BNA;
  const int tid = threadIdx.x, lane = tid & 31, wid = tid >> 5;
  const int wm = (wid & 3) * 32, wn = (wid >> 2) * 32;
  const float* Xe  = X  + (size_t)(e * TPE + mb) * HID;
  const float* W1e = W1 + (size_t)e * HID * (2 * INTER);
  const v8f vzero = {0.f, 0.f, 0.f, 0.f, 0.f, 0.f, 0.f, 0.f};
  v8f accG[2][2], accU[2][2];
#pragma unroll
  for (int mt = 0; mt < 2; ++mt)
#pragma unroll
    for (int nt = 0; nt < 2; ++nt) { accG[mt][nt] = vzero; accU[mt][nt] = vzero; }

  for (int k0 = 0; k0 < HID; k0 += KT) {
    __syncthreads();
#pragma unroll
    for (int i = 0; i < 2; ++i) {
      int c = tid + i * 256, row = c >> 2, kc = (c & 3) * 8;
      const float* src = Xe + (size_t)row * HID + k0 + kc;
      unsigned int pk[4];
#pragma unroll
      for (int j = 0; j < 4; ++j)
        pk[j] = (unsigned)f2bf(src[2 * j]) | ((unsigned)f2bf(src[2 * j + 1]) << 16);
      *(uint4*)&Al[row * KT + kc] = make_uint4(pk[0], pk[1], pk[2], pk[3]);
    }
#pragma unroll
    for (int i = 0; i < 2; ++i) {
      int c = tid + i * 256, n = c & 127, kc = (c >> 7) * 8;
      int col = (n < BNA) ? (f0 + n) : (INTER + f0 + (n - BNA));
      const float* src = W1e + (size_t)(k0 + kc) * (2 * INTER) + col;
      unsigned int pk[4];
#pragma unroll
      for (int j = 0; j < 4; ++j)
        pk[j] = (unsigned)f2bf(src[(size_t)(2 * j) * (2 * INTER)]) |
                ((unsigned)f2bf(src[(size_t)(2 * j + 1) * (2 * INTER)]) << 16);
      *(uint4*)&Bl[n * KT + kc] = make_uint4(pk[0], pk[1], pk[2], pk[3]);
    }
    __syncthreads();
    v16bf af[2];
    {
      int r15 = lane & 15, kb = (lane >> 4) * 16;
#pragma unroll
      for (int mt = 0; mt < 2; ++mt) {
        const unsigned short* p = &Al[(wm + mt * 16 + r15) * KT + kb];
        af[mt] = cat8(*(const v8bf*)p, *(const v8bf*)(p + 8));
      }
    }
    int kb2 = (lane >> 4) * 16;
#pragma unroll
    for (int nt = 0; nt < 2; ++nt) {
      int n = wn + nt * 16 + (lane & 15);
      v16bf bg = *(const v16bf*)&Bl[n * KT + kb2];
      v16bf bu = *(const v16bf*)&Bl[(BNA + n) * KT + kb2];
#pragma unroll
      for (int mt = 0; mt < 2; ++mt) {
        accG[mt][nt] = __builtin_amdgcn_wmma_f32_16x16x32_bf16(
            false, af[mt], false, bg, (short)0, accG[mt][nt], false, false);
        accU[mt][nt] = __builtin_amdgcn_wmma_f32_16x16x32_bf16(
            false, af[mt], false, bu, (short)0, accU[mt][nt], false, false);
      }
    }
  }
  unsigned short* Oe = acted + (size_t)e * TPE * INTER;
  const int rofs = (lane >> 4) << 3, c15 = lane & 15;
#pragma unroll
  for (int mt = 0; mt < 2; ++mt)
#pragma unroll
    for (int nt = 0; nt < 2; ++nt) {
      int row0 = mb + wm + mt * 16 + rofs;
      int col  = f0 + wn + nt * 16 + c15;
#pragma unroll
      for (int v = 0; v < 8; ++v)
        Oe[(size_t)(row0 + v) * INTER + col] =
            f2bf(accU[mt][nt][v] * silu(accG[mt][nt][v]));
    }
}

__global__ __launch_bounds__(256)
void k2_legacy(const unsigned short* __restrict__ acted, const float* __restrict__ W2,
               float* __restrict__ out) {
  constexpr int BM = 128, BN = 64, KT = 32;
  __shared__ alignas(32) unsigned short Al[BM * KT];
  __shared__ alignas(32) unsigned short Bl[BN * KT];
  const int e = blockIdx.z, mb = blockIdx.y * BM, n0 = blockIdx.x * BN;
  const int tid = threadIdx.x, lane = tid & 31, wid = tid >> 5;
  const int wm = (wid & 3) * 32, wn = (wid >> 2) * 32;
  const unsigned short* Ae = acted + (size_t)(e * TPE + mb) * INTER;
  const float* W2e = W2 + (size_t)e * INTER * HID;
  const v8f vzero = {0.f, 0.f, 0.f, 0.f, 0.f, 0.f, 0.f, 0.f};
  v8f acc[2][2];
#pragma unroll
  for (int mt = 0; mt < 2; ++mt)
#pragma unroll
    for (int nt = 0; nt < 2; ++nt) acc[mt][nt] = vzero;

  for (int k0 = 0; k0 < INTER; k0 += KT) {
    __syncthreads();
    {
      int row = tid >> 1, kc = (tid & 1) * 16;
      const uint4* src = (const uint4*)(Ae + (size_t)row * INTER + k0 + kc);
      uint4* dst = (uint4*)&Al[row * KT + kc];
      dst[0] = src[0]; dst[1] = src[1];
    }
    {
      int n = tid & 63, kc = (tid >> 6) * 8;
      const float* src = W2e + (size_t)(k0 + kc) * HID + n0 + n;
      unsigned int pk[4];
#pragma unroll
      for (int j = 0; j < 4; ++j)
        pk[j] = (unsigned)f2bf(src[(size_t)(2 * j) * HID]) |
                ((unsigned)f2bf(src[(size_t)(2 * j + 1) * HID]) << 16);
      *(uint4*)&Bl[n * KT + kc] = make_uint4(pk[0], pk[1], pk[2], pk[3]);
    }
    __syncthreads();
    v16bf af[2];
    {
      int r15 = lane & 15, kb = (lane >> 4) * 16;
#pragma unroll
      for (int mt = 0; mt < 2; ++mt) {
        const unsigned short* p = &Al[(wm + mt * 16 + r15) * KT + kb];
        af[mt] = cat8(*(const v8bf*)p, *(const v8bf*)(p + 8));
      }
    }
    int kb2 = (lane >> 4) * 16;
#pragma unroll
    for (int nt = 0; nt < 2; ++nt) {
      int n = wn + nt * 16 + (lane & 15);
      v16bf b = *(const v16bf*)&Bl[n * KT + kb2];
#pragma unroll
      for (int mt = 0; mt < 2; ++mt)
        acc[mt][nt] = __builtin_amdgcn_wmma_f32_16x16x32_bf16(
            false, af[mt], false, b, (short)0, acc[mt][nt], false, false);
    }
  }
  float* Oe = out + (size_t)e * TPE * HID;
  const int rofs = (lane >> 4) << 3, c15 = lane & 15;
#pragma unroll
  for (int mt = 0; mt < 2; ++mt)
#pragma unroll
    for (int nt = 0; nt < 2; ++nt) {
      int row0 = mb + wm + mt * 16 + rofs;
      int col  = n0 + wn + nt * 16 + c15;
#pragma unroll
      for (int v = 0; v < 8; ++v)
        Oe[(size_t)(row0 + v) * HID + col] = acc[mt][nt][v];
    }
}

// ---------------------------------------------------------------------------
extern "C" void kernel_launch(void* const* d_in, const int* in_sizes, int n_in,
                              void* d_out, int out_size, void* d_ws, size_t ws_size,
                              hipStream_t stream) {
  (void)in_sizes; (void)n_in; (void)out_size;
  const float* X  = (const float*)d_in[0];   // (8192, 2048)
  const float* W1 = (const float*)d_in[1];   // (8, 2048, 8192)
  const float* W2 = (const float*)d_in[2];   // (8, 4096, 2048)
  float* out = (float*)d_out;                // (8192, 2048)

  const size_t X_SZ  = (size_t)E_NUM * TPE * HID;          // 16.8M
  const size_t W1_SZ = (size_t)E_NUM * HID * 2 * INTER;    // 134.2M
  const size_t W2_SZ = (size_t)E_NUM * INTER * HID;        // 67.1M
  const size_t A_SZ  = (size_t)E_NUM * TPE * INTER;        // 33.6M
  const size_t need_fast = (X_SZ + W1_SZ + W2_SZ + A_SZ) * sizeof(unsigned short);

  dim3 blk(256);
  if (ws_size >= need_fast) {
    unsigned short* Xb    = (unsigned short*)d_ws;
    unsigned short* W1t   = Xb + X_SZ;
    unsigned short* W2t   = W1t + W1_SZ;
    unsigned short* acted = W2t + W2_SZ;

    cvt_flat<<<dim3((unsigned)(X_SZ / 8 / 256)), blk, 0, stream>>>(X, Xb);
    cvt_transpose<<<dim3(2 * INTER / 64, HID / 64, E_NUM), blk, 0, stream>>>(
        W1, W1t, HID, 2 * INTER);
    cvt_transpose<<<dim3(HID / 64, INTER / 64, E_NUM), blk, 0, stream>>>(
        W2, W2t, INTER, HID);

    k1_fast<<<dim3(INTER / 64, TPE / 128, E_NUM), blk, 0, stream>>>(Xb, W1t, acted);
    k2_fast<<<dim3(HID / 64, TPE / 128, E_NUM), blk, 0, stream>>>(acted, W2t, out);
  } else {
    unsigned short* acted = (unsigned short*)d_ws;   // 64 MiB
    k1_legacy<<<dim3(INTER / 64, TPE / 128, E_NUM), blk, 0, stream>>>(X, W1, acted);
    k2_legacy<<<dim3(HID / 64, TPE / 128, E_NUM), blk, 0, stream>>>(acted, W2, out);
  }
}
